// DecoderRNN_29557964931539
// MI455X (gfx1250) — matrix operand order repaired and checked by reading, not compiled, not run on hardware
//
#include <hip/hip_runtime.h>
#include <hip/hip_bf16.h>

// ---------------------------------------------------------------------------
// GRU decoder (B=16, L=128, H=512, V=32000) on gfx1250:
// bf16 WMMA GEMMs with async global->LDS B-tile staging (ASYNCcnt path).
// ---------------------------------------------------------------------------

typedef __attribute__((ext_vector_type(16))) __bf16 v16bf;
typedef __attribute__((ext_vector_type(8)))  float  v8f;

union BFrag { v16bf v; uint4 q[2]; };

#define WMMA_BF16(a, b, c) \
  __builtin_amdgcn_wmma_f32_16x16x32_bf16(false, (a), false, (b), (short)0, (c), false, false)

// A-fragment (16x32 bf16): lane m = lane&15; half 0 holds K {0..7,16..23},
// half 1 holds K {8..15,24..31}.  p points at row m, offset k0.
__device__ static inline v16bf load_a_frag(const __bf16* p, int hf) {
  BFrag f;
  f.q[0] = *(const uint4*)(p + 8 * hf);
  f.q[1] = *(const uint4*)(p + 16 + 8 * hf);
  return f.v;
}
// B-fragment (32x16 bf16): lane holds column n = lane&15; half 0 holds
// K {0..7,16..23}, half 1 holds K {8..15,24..31}.  p points at col n's K-run, offset k0.
__device__ static inline v16bf load_b_frag(const __bf16* p, int hf) {
  BFrag f;
  f.q[0] = *(const uint4*)(p + 8 * hf);
  f.q[1] = *(const uint4*)(p + 16 + 8 * hf);
  return f.v;
}

// ---------------------------------------------------------------------------
// f32 -> bf16 conversion
// ---------------------------------------------------------------------------
__global__ void f2bf_kernel(const float* __restrict__ in, __bf16* __restrict__ out, long long n) {
  long long i = (long long)blockIdx.x * blockDim.x + threadIdx.x;
  if (i < n) out[i] = (__bf16)in[i];
}

// ---------------------------------------------------------------------------
// Embedding gather with SOS shift: dec_in = [SOS, target[:, :L-1]]
// ---------------------------------------------------------------------------
__global__ void embed_kernel(const int* __restrict__ target, const float* __restrict__ emb,
                             __bf16* __restrict__ x) {
  int i = blockIdx.x;              // row = b*128 + t
  int b = i >> 7, t = i & 127;
  int tok = (t == 0) ? 1 : target[b * 128 + (t - 1)];   // SOS = 1
  const float* src = emb + (long long)tok * 512;
  for (int k = threadIdx.x; k < 512; k += blockDim.x)
    x[(long long)i * 512 + k] = (__bf16)src[k];
}

// ---------------------------------------------------------------------------
// GEMM: C[2048 x N] = A[2048 x 512] * W[N x 512]^T + bias[N], f32 out.
// Block tile: 128 rows x 64 cols (8 waves; wave w = rows w*16..w*16+15).
// B (64 cols x K=512) streamed through LDS in 8 K-chunks of 64, double
// buffered via gfx1250 async global->LDS b128 copies (ASYNCcnt).
// Grid: blocks = 16 * ngroups  (mgrp = 0..15, ng = 0..ngroups-1).
// ---------------------------------------------------------------------------
__global__ __launch_bounds__(256) void gemm_bf16_wmma(
    const __bf16* __restrict__ A, const __bf16* __restrict__ W,
    const float* __restrict__ bias, float* __restrict__ C,
    int N, int ngroups) {
  constexpr int K = 512;
  constexpr int KCH = 64;                 // K per staged chunk
  constexpr int NCHUNK = K / KCH;         // 8
  constexpr int BROW = KCH + 8;           // padded LDS row stride (elements)
  __shared__ __align__(16) __bf16 bstage[2][64 * BROW];   // ~18.4 KB

  int tid  = threadIdx.x;
  int wave = tid >> 5;
  int lane = tid & 31;
  int hf   = lane >> 4;
  int nn   = lane & 15;
  int mgrp = blockIdx.x / ngroups;
  int ng   = blockIdx.x % ngroups;
  int row0 = mgrp * 128 + wave * 16;

  // Stage one K-chunk of B into LDS: 64 cols x 64 k = 8 KB.
  // 256 threads x 2 async b128 copies (16 B / lane) cover it exactly.
  auto stage = [&](int chunk, int buf) {
    const __bf16* wbase = W + (long long)(ng * 64) * K + chunk * KCH;
#pragma unroll
    for (int q = 0; q < 2; ++q) {
      int e  = (q * 256 + tid) * 8;       // element index within chunk
      int c  = e >> 6;                    // column 0..63
      int kk = e & 63;                    // k offset within chunk (mult of 8)
      const __bf16* src = wbase + (long long)c * K + kk;
      unsigned dst = (unsigned)(size_t)&bstage[buf][c * BROW + kk]; // LDS byte offset
      asm volatile("global_load_async_to_lds_b128 %0, %1, off"
                   :: "v"(dst), "v"(src) : "memory");
    }
  };

  v8f acc[4];
#pragma unroll
  for (int j = 0; j < 4; ++j) {
    float bv = bias[ng * 64 + j * 16 + nn];
#pragma unroll
    for (int r = 0; r < 8; ++r) acc[j][r] = bv;
  }

  const __bf16* arow = A + (long long)(row0 + (lane & 15)) * K;

  stage(0, 0);
  for (int ch = 0; ch < NCHUNK; ++ch) {
    asm volatile("s_wait_asynccnt 0x0" ::: "memory");  // my staged copies done
    __syncthreads();                                    // everyone's copies done
    if (ch + 1 < NCHUNK) stage(ch + 1, (ch + 1) & 1);   // prefetch next chunk
    const __bf16* bbuf = &bstage[ch & 1][0];
#pragma unroll
    for (int ks = 0; ks < KCH; ks += 32) {
      v16bf a = load_a_frag(arow + ch * KCH + ks, hf);
#pragma unroll
      for (int j = 0; j < 4; ++j) {
        v16bf b = load_b_frag(bbuf + (j * 16 + nn) * BROW + ks, hf);
        acc[j] = WMMA_BF16(a, b, acc[j]);
      }
    }
    __syncthreads();                                    // done reading buf before restage
  }

#pragma unroll
  for (int j = 0; j < 4; ++j)
#pragma unroll
    for (int r = 0; r < 8; ++r)
      C[(long long)(row0 + r + 8 * hf) * N + ng * 64 + j * 16 + nn] = acc[j][r];
}

// ---------------------------------------------------------------------------
// GRU scan: one workgroup, 8 waves, 128 sequential steps.
// hp = h * W_hh^T + b_hh via WMMA (M = 16 = batch, one M-tile).
// Wave w owns hidden columns [w*64, w*64+64) across all 3 gates.
// ---------------------------------------------------------------------------
__global__ __launch_bounds__(256) void gru_scan_kernel(
    const float* __restrict__ h0, const float* __restrict__ xp,
    const __bf16* __restrict__ Whh, const float* __restrict__ b_hh,
    __bf16* __restrict__ hs, float* __restrict__ h_last) {
  __shared__ __align__(16) __bf16 h_bf[16 * 512];   // 16 KB  (A-matrix source)
  __shared__ float  h_f [16 * 512];                 // 32 KB  (exact h for z*h)

  const int K = 512;
  int tid  = threadIdx.x;
  int wave = tid >> 5;
  int lane = tid & 31;
  int hf   = lane >> 4;
  int coln = lane & 15;
  int cbase = wave * 4;               // first of 4 owned column tiles

  for (int i = tid; i < 16 * 512; i += 256) {
    float v = h0[i];
    h_f[i] = v;
    h_bf[i] = (__bf16)v;
  }
  __syncthreads();

  for (int t = 0; t < 128; ++t) {
    v8f acc[3][4];
#pragma unroll
    for (int g = 0; g < 3; ++g)
#pragma unroll
      for (int j = 0; j < 4; ++j) {
        float bv = b_hh[g * 512 + (cbase + j) * 16 + coln];
#pragma unroll
        for (int r = 0; r < 8; ++r) acc[g][j][r] = bv;
      }

    // hp = h @ W_hh^T  (A from LDS, B from global/L2)
    for (int k0 = 0; k0 < K; k0 += 32) {
      v16bf a = load_a_frag(&h_bf[(lane & 15) * 512 + k0], hf);
#pragma unroll
      for (int g = 0; g < 3; ++g)
#pragma unroll
        for (int j = 0; j < 4; ++j) {
          const __bf16* brow = Whh + (long long)(g * 512 + (cbase + j) * 16 + coln) * K + k0;
          v16bf b = load_b_frag(brow, hf);
          acc[g][j] = WMMA_BF16(a, b, acc[g][j]);
        }
    }
    __syncthreads();   // all A reads of h_bf done before h is rewritten

    // gates + state update for the 16x64 slice this wave owns
#pragma unroll
    for (int j = 0; j < 4; ++j) {
      int col = (cbase + j) * 16 + coln;
#pragma unroll
      for (int r = 0; r < 8; ++r) {
        int m = r + 8 * hf;                       // batch index
        long long xrow = ((long long)m * 128 + t) * 1536;
        float xr = xp[xrow + col];
        float xz = xp[xrow + 512 + col];
        float xn = xp[xrow + 1024 + col];
        float hr = acc[0][j][r], hz = acc[1][j][r], hn = acc[2][j][r];
        float rg = 1.0f / (1.0f + __expf(-(xr + hr)));
        float zg = 1.0f / (1.0f + __expf(-(xz + hz)));
        float ngate = tanhf(xn + rg * hn);
        float hold = h_f[m * 512 + col];
        float hnew = (1.0f - zg) * ngate + zg * hold;
        h_f[m * 512 + col] = hnew;
        h_bf[m * 512 + col] = (__bf16)hnew;
        hs[((long long)m * 128 + t) * 512 + col] = (__bf16)hnew;
        if (t == 127) h_last[m * 512 + col] = hnew;
      }
    }
    __syncthreads();   // h updated before next step's A loads
  }
}

// ---------------------------------------------------------------------------
// Row-wise logsumexp over V = 32000 (one block per row)
// ---------------------------------------------------------------------------
__global__ __launch_bounds__(256) void row_lse_kernel(const float* __restrict__ logits,
                                                      float* __restrict__ lse) {
  __shared__ float red[256];
  int row = blockIdx.x;
  const float* p = logits + (long long)row * 32000;

  float m = -1e30f;
  for (int v = threadIdx.x; v < 32000; v += 256) m = fmaxf(m, p[v]);
  red[threadIdx.x] = m;
  __syncthreads();
  for (int s = 128; s > 0; s >>= 1) {
    if (threadIdx.x < s) red[threadIdx.x] = fmaxf(red[threadIdx.x], red[threadIdx.x + s]);
    __syncthreads();
  }
  m = red[0];
  __syncthreads();

  float sum = 0.0f;
  for (int v = threadIdx.x; v < 32000; v += 256) sum += __expf(p[v] - m);
  red[threadIdx.x] = sum;
  __syncthreads();
  for (int s = 128; s > 0; s >>= 1) {
    if (threadIdx.x < s) red[threadIdx.x] += red[threadIdx.x + s];
    __syncthreads();
  }
  if (threadIdx.x == 0) lse[row] = m + __logf(red[0]);
}

__global__ void sub_lse_kernel(float* __restrict__ logits, const float* __restrict__ lse) {
  long long i = (long long)blockIdx.x * blockDim.x + threadIdx.x;
  if (i < (long long)2048 * 32000) logits[i] -= lse[i / 32000];
}

// ---------------------------------------------------------------------------
extern "C" void kernel_launch(void* const* d_in, const int* in_sizes, int n_in,
                              void* d_out, int out_size, void* d_ws, size_t ws_size,
                              hipStream_t stream) {
  (void)in_sizes; (void)n_in; (void)out_size; (void)ws_size;

  const float* enc_hid = (const float*)d_in[1];   // (1,16,512)  -> h0
  const int*   target  = (const int*)d_in[2];     // (16,128)
  const float* emb     = (const float*)d_in[3];   // (32000,512)
  const float* W_ih    = (const float*)d_in[4];   // (1536,512)
  const float* W_hh    = (const float*)d_in[5];   // (1536,512)
  const float* b_ih    = (const float*)d_in[6];   // (1536)
  const float* b_hh    = (const float*)d_in[7];   // (1536)
  const float* W_out   = (const float*)d_in[8];   // (32000,512)
  const float* b_out   = (const float*)d_in[9];   // (32000)
  float* out = (float*)d_out;                     // 2048*32000 logprobs + 16*512 h_last

  // workspace layout (bytes)
  char* ws = (char*)d_ws;
  __bf16* W_ih_bf  = (__bf16*)(ws + 0);                       //  1,572,864
  __bf16* W_hh_bf  = (__bf16*)(ws + 1572864);                 //  1,572,864
  __bf16* W_out_bf = (__bf16*)(ws + 3145728);                 // 32,768,000
  __bf16* x_bf     = (__bf16*)(ws + 35913728);                //  2,097,152
  float*  xp       = (float*) (ws + 38010880);                // 12,582,912
  __bf16* hs_bf    = (__bf16*)(ws + 50593792);                //  2,097,152
  float*  lse      = (float*) (ws + 52690944);                //      8,192

  // 1) weight downconversion to bf16
  f2bf_kernel<<<(786432 + 255) / 256, 256, 0, stream>>>(W_ih, W_ih_bf, 786432LL);
  f2bf_kernel<<<(786432 + 255) / 256, 256, 0, stream>>>(W_hh, W_hh_bf, 786432LL);
  f2bf_kernel<<<(16384000 + 255) / 256, 256, 0, stream>>>(W_out, W_out_bf, 16384000LL);

  // 2) embedding gather (teacher forcing with SOS shift)
  embed_kernel<<<2048, 256, 0, stream>>>(target, emb, x_bf);

  // 3) input projection: xp = x @ W_ih^T + b_ih   (2048 x 1536)
  //    grid = 16 M-groups * 24 N-groups
  gemm_bf16_wmma<<<16 * 24, 256, 0, stream>>>(x_bf, W_ih_bf, b_ih, xp, 1536, 24);

  // 4) sequential GRU scan (single workgroup, WMMA recurrent matmul)
  gru_scan_kernel<<<1, 256, 0, stream>>>(enc_hid, xp, W_hh_bf, b_hh, hs_bf,
                                         out + (long long)2048 * 32000);

  // 5) logits = hs @ W_out^T + b_out  (2048 x 32000), straight into d_out
  //    grid = 16 M-groups * 500 N-groups
  gemm_bf16_wmma<<<16 * 500, 256, 0, stream>>>(hs_bf, W_out_bf, b_out, out, 32000, 500);

  // 6) log_softmax in place
  row_lse_kernel<<<2048, 256, 0, stream>>>(out, lse);
  sub_lse_kernel<<<(int)(((long long)2048 * 32000 + 255) / 256), 256, 0, stream>>>(out, lse);
}
